// NeRF_46892452938411
// MI455X (gfx1250) — matrix operand order repaired and checked by reading, not compiled, not run on hardware
//
#include <hip/hip_runtime.h>

typedef _Float16 v16h __attribute__((ext_vector_type(16)));
typedef _Float16 v8h  __attribute__((ext_vector_type(8)));
typedef __fp16   v2hf __attribute__((ext_vector_type(2)));
typedef float    v8f  __attribute__((ext_vector_type(8)));

#define N_ENC 16384
#define WPB   8
#define TPB   256
// ds_swizzle group-of-32: xor_mask=0x10, or=0, and=0x1f  -> swap lanes n <-> n+16
#define SWZ_SWAPX16 0x401F

__device__ __forceinline__ v8f wmma16(v16h a, v16h b, v8f c) {
    return __builtin_amdgcn_wmma_f32_16x16x32_f16(false, a, false, b, (short)0, c, false, false);
}

// single v_cvt_pk_rtz_f16_f32
__device__ __forceinline__ unsigned pk2(float a, float b) {
    v2hf h = __builtin_amdgcn_cvt_pkrtz(a, b);
    return __builtin_bit_cast(unsigned, h);
}
// pack + fused packed relu (v_pk_max_num_f16)
__device__ __forceinline__ unsigned pk2r(float a, float b) {
    v2hf h = __builtin_amdgcn_cvt_pkrtz(a, b);
    v2hf z = {(__fp16)0.f, (__fp16)0.f};
    h = __builtin_elementwise_max(h, z);
    return __builtin_bit_cast(unsigned, h);
}

__device__ __forceinline__ v16h cat8(v8h lo, v8h hi) {
    return __builtin_shufflevector(lo, hi, 0,1,2,3,4,5,6,7,8,9,10,11,12,13,14,15);
}

// Weight A fragment: 16x32 f16 tile of W^T (LDS wT[m][k], row stride ldk halfs).
// A layout: lanes 0-15 (M=lane): K=kb..kb+15 (v0-7); lanes 16-31: +16.
__device__ __forceinline__ v16h load_fragW(const _Float16* wT, int lane, int mb, int kb, int ldk) {
    const int m = lane & 15, hs = lane >> 4;
    const _Float16* p = wT + (mb + m) * ldk + kb + hs * 16;
    v8h lo = *reinterpret_cast<const v8h*>(p);
    v8h hi = *reinterpret_cast<const v8h*>(p + 8);
    return cat8(lo, hi);
}

// Bias preloads C: D row M = mb + r + 8*hs  ->  8 consecutive floats per half-wave.
__device__ __forceinline__ v8f load_bias(const float* b, int lane, int mb) {
    const float* p = b + mb + (lane >> 4) * 8;
    v8f c;
    #pragma unroll
    for (int r = 0; r < 8; ++r) c[r] = p[r];
    return c;
}

// 4 D tiles (features mt*16 + r + 8*hs, points across lanes) -> two B fragments
// (K=0..31, K=32..63) for the next transposed layer. Half-wave swap via ds_swizzle.
__device__ __forceinline__ void accs_to_bfrags(const v8f* acc, bool vlo, bool relu,
                                               v16h& b0, v16h& b1) {
    union { v16h h; unsigned u[8]; } r0, r1;
    #pragma unroll
    for (int d = 0; d < 4; ++d) {
        const unsigned p0 = relu ? pk2r(acc[0][2*d], acc[0][2*d+1]) : pk2(acc[0][2*d], acc[0][2*d+1]);
        const unsigned p1 = relu ? pk2r(acc[1][2*d], acc[1][2*d+1]) : pk2(acc[1][2*d], acc[1][2*d+1]);
        const unsigned p2 = relu ? pk2r(acc[2][2*d], acc[2][2*d+1]) : pk2(acc[2][2*d], acc[2][2*d+1]);
        const unsigned p3 = relu ? pk2r(acc[3][2*d], acc[3][2*d+1]) : pk2(acc[3][2*d], acc[3][2*d+1]);
        const unsigned q0 = (unsigned)__builtin_amdgcn_ds_swizzle((int)p0, SWZ_SWAPX16);
        const unsigned q1 = (unsigned)__builtin_amdgcn_ds_swizzle((int)p1, SWZ_SWAPX16);
        const unsigned q2 = (unsigned)__builtin_amdgcn_ds_swizzle((int)p2, SWZ_SWAPX16);
        const unsigned q3 = (unsigned)__builtin_amdgcn_ds_swizzle((int)p3, SWZ_SWAPX16);
        r0.u[d]     = vlo ? p0 : q1;   // K lo-8 of this half
        r0.u[d + 4] = vlo ? q0 : p1;   // K hi-8 of this half
        r1.u[d]     = vlo ? p2 : q3;
        r1.u[d + 4] = vlo ? q2 : p3;
    }
    b0 = r0.h; b1 = r1.h;
}

// 16-feature D tile -> single B fragment, K padded 16->32 with zeros (no relu).
__device__ __forceinline__ v16h acc_to_bfrag16(const v8f& a, bool vlo) {
    union { v16h h; unsigned u[8]; } r;
    #pragma unroll
    for (int d = 0; d < 4; ++d) {
        const unsigned p = pk2(a[2*d], a[2*d+1]);
        const unsigned q = (unsigned)__builtin_amdgcn_ds_swizzle((int)p, SWZ_SWAPX16);
        r.u[d]     = vlo ? p : 0u;
        r.u[d + 4] = vlo ? q : 0u;
    }
    return r.h;
}

// fout=64 layer: 4 M-tiles, NK K-chunks. Weights from LDS as A; activations as B in regs.
template<int LDK, int NK>
__device__ __forceinline__ void layer4(const _Float16* wT, const float* bias, int lane,
                                       const v16h* bk, v8f* acc) {
    #pragma unroll
    for (int mt = 0; mt < 4; ++mt) {
        v8f c = load_bias(bias, lane, mt * 16);
        #pragma unroll
        for (int kc = 0; kc < NK; ++kc)
            c = wmma16(load_fragW(wT, lane, mt * 16, kc * 32, LDK), bk[kc], c);
        acc[mt] = c;
    }
}

// fout=16 layer: single M-tile.
template<int LDK, int NK>
__device__ __forceinline__ v8f layer1t(const _Float16* wT, const float* bias, int lane,
                                       const v16h* bk) {
    v8f c = load_bias(bias, lane, 0);
    #pragma unroll
    for (int kc = 0; kc < NK; ++kc)
        c = wmma16(load_fragW(wT, lane, 0, kc * 32, LDK), bk[kc], c);
    return c;
}

__global__ __launch_bounds__(TPB) void nerf_fused_kernel(
    const float* __restrict__ x, const float* __restrict__ tables,
    const float* __restrict__ w1,  const float* __restrict__ b1,
    const float* __restrict__ w2,  const float* __restrict__ b2,
    const float* __restrict__ w3,  const float* __restrict__ b3,
    const float* __restrict__ cw1, const float* __restrict__ cb1,
    const float* __restrict__ cw2, const float* __restrict__ cb2,
    const float* __restrict__ cw3, const float* __restrict__ cb3,
    const float* __restrict__ cw4, const float* __restrict__ cb4,
    float* __restrict__ sigma_out, float* __restrict__ c_out, int B)
{
    // f16 W^T in LDS: wT[m][k] = W[k][m]; K / M padded where needed.
    __shared__ __align__(16) _Float16 s_w1t [64 * 32];
    __shared__ __align__(16) _Float16 s_w2t [64 * 64];
    __shared__ __align__(16) _Float16 s_w3t [16 * 64];
    __shared__ __align__(16) _Float16 s_cw1t[64 * 32];  // K padded 16->32
    __shared__ __align__(16) _Float16 s_cw2t[64 * 64];
    __shared__ __align__(16) _Float16 s_cw3t[64 * 64];
    __shared__ __align__(16) _Float16 s_cw4t[16 * 64];  // M padded 3->16
    __shared__ __align__(16) float s_b1[64], s_b2[64], s_b3[16];
    __shared__ __align__(16) float s_cb1[64], s_cb2[64], s_cb3[64], s_cb4[16];

    const int tid = threadIdx.x;

    for (int i = tid; i < 64 * 32; i += TPB) { int m = i >> 5, k = i & 31; s_w1t[i] = (_Float16)w1[k * 64 + m]; }
    for (int i = tid; i < 64 * 64; i += TPB) { int m = i >> 6, k = i & 63; s_w2t[i] = (_Float16)w2[k * 64 + m]; }
    for (int i = tid; i < 16 * 64; i += TPB) { int m = i >> 6, k = i & 63; s_w3t[i] = (_Float16)w3[k * 16 + m]; }
    for (int i = tid; i < 64 * 32; i += TPB) { int m = i >> 5, k = i & 31; s_cw1t[i] = (k < 16) ? (_Float16)cw1[k * 64 + m] : (_Float16)0.f; }
    for (int i = tid; i < 64 * 64; i += TPB) { int m = i >> 6, k = i & 63; s_cw2t[i] = (_Float16)cw2[k * 64 + m]; }
    for (int i = tid; i < 64 * 64; i += TPB) { int m = i >> 6, k = i & 63; s_cw3t[i] = (_Float16)cw3[k * 64 + m]; }
    for (int i = tid; i < 16 * 64; i += TPB) { int m = i >> 6, k = i & 63; s_cw4t[i] = (m < 3) ? (_Float16)cw4[k * 3 + m] : (_Float16)0.f; }
    if (tid < 64) { s_b1[tid] = b1[tid]; s_b2[tid] = b2[tid]; s_cb1[tid] = cb1[tid]; s_cb2[tid] = cb2[tid]; s_cb3[tid] = cb3[tid]; }
    if (tid < 16) { s_b3[tid] = b3[tid]; s_cb4[tid] = (tid < 3) ? cb4[tid] : 0.f; }
    __syncthreads();

    const int lane = tid & 31;
    const int wid  = tid >> 5;
    const int n    = lane & 15;       // point within tile (B-operand column)
    const int hs   = lane >> 4;       // feature half
    const bool vlo = (hs == 0);
    const int gwave  = blockIdx.x * WPB + wid;
    const int nwaves = gridDim.x * WPB;
    const int ntiles = B >> 4;
    const char* tbl = (const char*)tables;

    for (int t = gwave; t < ntiles; t += nwaves) {
        const int base = t << 4;

        // ===== hash encode -> B fragment directly in registers =====
        // lane n holds features K = hs*16 .. hs*16+15 of point n  (level l = hs*8+li).
        union { v16h h; unsigned u[8]; } be;
        {
            const int gp = base + n;
            const float px = x[gp * 3 + 0], py = x[gp * 3 + 1], pz = x[gp * 3 + 2];
            const int RES[16] = {16,20,25,32,40,50,64,80,101,128,161,203,256,322,406,512};
            #pragma unroll
            for (int li = 0; li < 8; ++li) {
                const int res = vlo ? RES[li] : RES[li + 8];
                const unsigned lvbase = ((unsigned)(hs * 8 + li)) << 17;  // l * 16384 * 8 bytes
                const float sc = 0.5f * (float)(res - 1);
                const float gx = (px + 1.f) * sc, gy = (py + 1.f) * sc, gz = (pz + 1.f) * sc;
                const float fx = floorf(gx), fy = floorf(gy), fz = floorf(gz);
                const float wx = gx - fx, wy = gy - fy, wz = gz - fz;
                int ix0 = (int)fx; ix0 = ix0 < 0 ? 0 : (ix0 > res - 1 ? res - 1 : ix0);
                int iy0 = (int)fy; iy0 = iy0 < 0 ? 0 : (iy0 > res - 1 ? res - 1 : iy0);
                int iz0 = (int)fz; iz0 = iz0 < 0 ? 0 : (iz0 > res - 1 ? res - 1 : iz0);
                const int ix1 = ix0 + 1 > res - 1 ? res - 1 : ix0 + 1;
                const int iy1 = iy0 + 1 > res - 1 ? res - 1 : iy0 + 1;
                const int iz1 = iz0 + 1 > res - 1 ? res - 1 : iz0 + 1;
                float f0 = 0.f, f1 = 0.f;
                #pragma unroll
                for (int c = 0; c < 8; ++c) {
                    const int cx = c & 1, cy = (c >> 1) & 1, cz = c >> 2;
                    const int ix = cx ? ix1 : ix0, iy = cy ? iy1 : iy0, iz = cz ? iz1 : iz0;
                    const unsigned hidx = ((unsigned)ix ^ ((unsigned)iy * 2654435761u)
                                           ^ ((unsigned)iz * 805459861u)) & (unsigned)(N_ENC - 1);
                    unsigned idx;
                    if (li < 3) {   // only levels 0..2 (hs==0) are dense
                        const unsigned didx = (unsigned)(ix + res * (iy + res * iz));
                        idx = vlo ? didx : hidx;
                    } else {
                        idx = hidx;
                    }
                    const float wc = (cx ? wx : 1.f - wx) * (cy ? wy : 1.f - wy) * (cz ? wz : 1.f - wz);
                    const float2 e = *reinterpret_cast<const float2*>(tbl + (lvbase + idx * 8u));
                    f0 += wc * e.x;
                    f1 += wc * e.y;
                }
                be.u[li] = pk2(f0, f1);
            }
        }
        __builtin_amdgcn_sched_barrier(0);   // fence encoder from MLP (bound reg pressure)

        v8f acc[4];
        v16h bk[2];
        bk[0] = be.h;

        // density L1: 32 -> 64, relu
        layer4<32, 1>(s_w1t, s_b1, lane, bk, acc);
        accs_to_bfrags(acc, vlo, true, bk[0], bk[1]);
        __builtin_amdgcn_sched_barrier(0);
        // density L2: 64 -> 64, relu
        layer4<64, 2>(s_w2t, s_b2, lane, bk, acc);
        accs_to_bfrags(acc, vlo, true, bk[0], bk[1]);
        __builtin_amdgcn_sched_barrier(0);
        // density L3: 64 -> 16, no relu; sigma = exp(d[:,0])
        v8f d = layer1t<64, 2>(s_w3t, s_b3, lane, bk);
        if (vlo) sigma_out[base + n] = __expf(d[0]);
        bk[0] = acc_to_bfrag16(d, vlo);
        __builtin_amdgcn_sched_barrier(0);
        // color L1: 16(->32) -> 64, relu
        layer4<32, 1>(s_cw1t, s_cb1, lane, bk, acc);
        accs_to_bfrags(acc, vlo, true, bk[0], bk[1]);
        __builtin_amdgcn_sched_barrier(0);
        // color L2: 64 -> 64, relu
        layer4<64, 2>(s_cw2t, s_cb2, lane, bk, acc);
        accs_to_bfrags(acc, vlo, true, bk[0], bk[1]);
        __builtin_amdgcn_sched_barrier(0);
        // color L3: 64 -> 64, relu
        layer4<64, 2>(s_cw3t, s_cb3, lane, bk, acc);
        accs_to_bfrags(acc, vlo, true, bk[0], bk[1]);
        __builtin_amdgcn_sched_barrier(0);
        // color out: 64 -> 3(->16), sigmoid; channels are rows M=0..2 -> lanes 0-15 hold all 3
        v8f co = layer1t<64, 2>(s_cw4t, s_cb4, lane, bk);
        if (vlo) {
            float3 rgb;
            rgb.x = 1.f / (1.f + __expf(-co[0]));
            rgb.y = 1.f / (1.f + __expf(-co[1]));
            rgb.z = 1.f / (1.f + __expf(-co[2]));
            *reinterpret_cast<float3*>(&c_out[(size_t)(base + n) * 3]) = rgb;
        }
        __builtin_amdgcn_sched_barrier(0);
    }
}

extern "C" void kernel_launch(void* const* d_in, const int* in_sizes, int n_in,
                              void* d_out, int out_size, void* d_ws, size_t ws_size,
                              hipStream_t stream) {
    const float* x      = (const float*)d_in[0];
    const float* tables = (const float*)d_in[1];
    const float* w1  = (const float*)d_in[2];
    const float* b1  = (const float*)d_in[3];
    const float* w2  = (const float*)d_in[4];
    const float* b2  = (const float*)d_in[5];
    const float* w3  = (const float*)d_in[6];
    const float* b3  = (const float*)d_in[7];
    const float* cw1 = (const float*)d_in[8];
    const float* cb1 = (const float*)d_in[9];
    const float* cw2 = (const float*)d_in[10];
    const float* cb2 = (const float*)d_in[11];
    const float* cw3 = (const float*)d_in[12];
    const float* cb3 = (const float*)d_in[13];
    const float* cw4 = (const float*)d_in[14];
    const float* cb4 = (const float*)d_in[15];

    const int B = in_sizes[0] / 3;
    float* sigma_out = (float*)d_out;
    float* c_out     = sigma_out + B;

    const int grid = 512;  // persistent blocks; 8 waves each grid-stride over B/16 tiles
    nerf_fused_kernel<<<grid, TPB, 0, stream>>>(
        x, tables, w1, b1, w2, b2, w3, b3,
        cw1, cb1, cw2, cb2, cw3, cb3, cw4, cb4,
        sigma_out, c_out, B);
}
